// RotaryCrossAttention_34239479284266
// MI455X (gfx1250) — matrix operand order repaired and checked by reading, not compiled, not run on hardware
//
#include <hip/hip_runtime.h>

typedef __attribute__((ext_vector_type(16))) _Float16 v16h;
typedef __attribute__((ext_vector_type(8)))  _Float16 v8h;
typedef __attribute__((ext_vector_type(4)))  _Float16 v4h;
typedef __attribute__((ext_vector_type(8)))  float    v8f;

#define NQ_TOK   8192      // B*Q_LEN
#define NKV_TOK  16384     // B*KV_LEN
#define DIMM     512
#define INNERD   512
#define NHEAD    8
#define DH       64
#define QL       512
#define KVL      1024

// ---------------------------------------------------------------------------
// WMMA helper: D = A(16x32 f16) * B(32x16 f16) + C(16x16 f32)
// ---------------------------------------------------------------------------
static __device__ inline v8f wmma16(v16h a, v16h b, v8f c) {
  return __builtin_amdgcn_wmma_f32_16x16x32_f16(false, a, false, b, (short)0, c,
                                                false, false);
}

// A-fragment (16x32 f16, rows striped per lane).  Per ISA 7.12.2:
// lane<16: row=lane, K = {kb..kb+7, 16+kb..16+kb+7} with kb=0
// lane>=16: row=lane-16, kb=8
static __device__ inline v16h load_a_frag(const _Float16* __restrict__ p, int ld,
                                          int row_base, int koff) {
  int lane = threadIdx.x & 31;
  int row  = row_base + (lane & 15);
  int kb   = koff + ((lane >> 4) << 3);
  const _Float16* q = p + (size_t)row * ld + kb;
  v8h lo = *(const v8h*)q;         // K = kb .. kb+7
  v8h hi = *(const v8h*)(q + 16);  // K = kb+16 .. kb+23
  v16h r;
#pragma unroll
  for (int t = 0; t < 8; ++t) { r[t] = lo[t]; r[t + 8] = hi[t]; }
  return r;
}

// B-fragment (32x16 f16, cols striped per lane).  Per ISA 7.12.x:
// lane<16: col=lane, K = {k0..k0+7, k0+16..k0+23} (k0=koff); lane>=16: k0=koff+8.
// Source stored column-major as [col][k].
static __device__ inline v16h load_b_frag(const _Float16* __restrict__ p, int ld,
                                          int col_base, int koff) {
  int lane = threadIdx.x & 31;
  int col  = col_base + (lane & 15);
  int k0   = koff + ((lane >> 4) << 3);
  const _Float16* q = p + (size_t)col * ld + k0;
  v8h lo = *(const v8h*)q;
  v8h hi = *(const v8h*)(q + 16);
  v16h r;
#pragma unroll
  for (int t = 0; t < 8; ++t) { r[t] = lo[t]; r[t + 8] = hi[t]; }
  return r;
}

// 64x64 wave-tile GEMM core: C[4][4] += A[row0..+64, :kdim] * B^T[col0..+64, :kdim]
static __device__ inline void gemm_core(const _Float16* __restrict__ A,
                                        const _Float16* __restrict__ Bt,
                                        int row0, int col0, int kdim,
                                        v8f c[4][4]) {
  for (int k = 0; k < kdim; k += 32) {
    v16h a[4], b[4];
#pragma unroll
    for (int i = 0; i < 4; ++i) a[i] = load_a_frag(A, kdim, row0 + 16 * i, k);
#pragma unroll
    for (int j = 0; j < 4; ++j) b[j] = load_b_frag(Bt, kdim, col0 + 16 * j, k);
#pragma unroll
    for (int i = 0; i < 4; ++i)
#pragma unroll
      for (int j = 0; j < 4; ++j) c[i][j] = wmma16(a[i], b[j], c[i][j]);
  }
}

// ---------------------------------------------------------------------------
// 1) LayerNorm (f32 in) -> f16 out.  One wave per 512-wide token row.
// ---------------------------------------------------------------------------
__global__ __launch_bounds__(256) void ln_f16_kernel(
    const float* __restrict__ x, const float* __restrict__ g,
    const float* __restrict__ b, _Float16* __restrict__ o, int ntok) {
  int wid = (blockIdx.x * blockDim.x + threadIdx.x) >> 5;
  if (wid >= ntok) return;
  int lane = threadIdx.x & 31;
  const float4* xr = (const float4*)(x + (size_t)wid * DIMM);
  float4 v[4];
  float s = 0.f;
#pragma unroll
  for (int i = 0; i < 4; ++i) {
    v[i] = xr[i * 32 + lane];
    s += v[i].x + v[i].y + v[i].z + v[i].w;
  }
#pragma unroll
  for (int off = 16; off > 0; off >>= 1) s += __shfl_xor(s, off, 32);
  float mu = s * (1.0f / 512.0f);
  float var = 0.f;
#pragma unroll
  for (int i = 0; i < 4; ++i) {
    float dx;
    dx = v[i].x - mu; var += dx * dx;
    dx = v[i].y - mu; var += dx * dx;
    dx = v[i].z - mu; var += dx * dx;
    dx = v[i].w - mu; var += dx * dx;
  }
#pragma unroll
  for (int off = 16; off > 0; off >>= 1) var += __shfl_xor(var, off, 32);
  float rstd = rsqrtf(var * (1.0f / 512.0f) + 1e-5f);
  const float4* g4 = (const float4*)g;
  const float4* b4 = (const float4*)b;
#pragma unroll
  for (int i = 0; i < 4; ++i) {
    float4 gg = g4[i * 32 + lane], bb = b4[i * 32 + lane];
    v4h h;
    h[0] = (_Float16)((v[i].x - mu) * rstd * gg.x + bb.x);
    h[1] = (_Float16)((v[i].y - mu) * rstd * gg.y + bb.y);
    h[2] = (_Float16)((v[i].z - mu) * rstd * gg.z + bb.z);
    h[3] = (_Float16)((v[i].w - mu) * rstd * gg.w + bb.w);
    *(v4h*)(o + (size_t)wid * DIMM + (i * 32 + lane) * 4) = h;
  }
}

// ---------------------------------------------------------------------------
// 2) Weight f32 [KD][ND] -> f16 transposed [ND][KD]
// ---------------------------------------------------------------------------
__global__ __launch_bounds__(256) void wt_f16_kernel(
    const float* __restrict__ W, _Float16* __restrict__ Wt, int ND, int KD) {
  int idx = blockIdx.x * 256 + threadIdx.x;
  int n = idx / KD, k = idx - n * KD;
  Wt[idx] = (_Float16)W[(size_t)k * ND + n];
}

// ---------------------------------------------------------------------------
// 3) Q GEMM + fused RoPE, scatter to Qh[b][h][q][d]
// ---------------------------------------------------------------------------
__global__ __launch_bounds__(128) void gemm_q_kernel(
    const _Float16* __restrict__ A, const _Float16* __restrict__ Bt,
    const float* __restrict__ ropeQ, _Float16* __restrict__ Qh) {
  int warp = threadIdx.x >> 5, lane = threadIdx.x & 31;
  int wid = blockIdx.x * 4 + warp;
  int row0 = (wid >> 3) * 64, col0 = (wid & 7) * 64;
  v8f c[4][4];
  const v8f vz = {};
#pragma unroll
  for (int i = 0; i < 4; ++i)
#pragma unroll
    for (int j = 0; j < 4; ++j) c[i][j] = vz;
  gemm_core(A, Bt, row0, col0, DIMM, c);

#pragma unroll
  for (int mf = 0; mf < 4; ++mf)
#pragma unroll
    for (int nf = 0; nf < 4; ++nf) {
      int n = col0 + nf * 16 + (lane & 15);
      int h = n >> 6, d = n & 63;
      int r0 = row0 + mf * 16 + ((lane >> 4) << 3);
      bool odd = d & 1;
#pragma unroll
      for (int i = 0; i < 8; ++i) {
        int row = r0 + i;
        float x = c[mf][nf][i];
        float part = __shfl_xor(x, 1, 32);
        float f = ropeQ[(size_t)row * DH + d];
        float cs = __cosf(f), sn = __sinf(f);
        float val = odd ? (x * cs + part * sn) : (x * cs - part * sn);
        int bq = row >> 9, q = row & (QL - 1);
        Qh[(((size_t)(bq * NHEAD + h) * QL) + q) * DH + d] = (_Float16)val;
      }
    }
}

// ---------------------------------------------------------------------------
// 4) KV GEMM + fused RoPE.  K -> Kh[b][h][kv][d], V -> Vt[b][h][d][kv]
// ---------------------------------------------------------------------------
__global__ __launch_bounds__(128) void gemm_kv_kernel(
    const _Float16* __restrict__ A, const _Float16* __restrict__ Bt,
    const float* __restrict__ ropeC, _Float16* __restrict__ Kh,
    _Float16* __restrict__ Vt) {
  int warp = threadIdx.x >> 5, lane = threadIdx.x & 31;
  int wid = blockIdx.x * 4 + warp;
  int row0 = (wid >> 4) * 64, col0 = (wid & 15) * 64;
  v8f c[4][4];
  const v8f vz = {};
#pragma unroll
  for (int i = 0; i < 4; ++i)
#pragma unroll
    for (int j = 0; j < 4; ++j) c[i][j] = vz;
  gemm_core(A, Bt, row0, col0, DIMM, c);

  bool isV = (col0 >= INNERD);
#pragma unroll
  for (int mf = 0; mf < 4; ++mf)
#pragma unroll
    for (int nf = 0; nf < 4; ++nf) {
      int n = col0 + nf * 16 + (lane & 15);
      int d = n & 63;
      int r0 = row0 + mf * 16 + ((lane >> 4) << 3);
      bool odd = d & 1;
      if (!isV) {
        int h = n >> 6;
#pragma unroll
        for (int i = 0; i < 8; ++i) {
          int row = r0 + i;
          float x = c[mf][nf][i];
          float part = __shfl_xor(x, 1, 32);
          float f = ropeC[(size_t)row * DH + d];
          float cs = __cosf(f), sn = __sinf(f);
          float val = odd ? (x * cs + part * sn) : (x * cs - part * sn);
          int bb = row >> 10, kvp = row & (KVL - 1);
          Kh[(((size_t)(bb * NHEAD + h) * KVL) + kvp) * DH + d] = (_Float16)val;
        }
      } else {
        int h = (n - INNERD) >> 6;
        v8h pk;
#pragma unroll
        for (int i = 0; i < 8; ++i) {
          int row = r0 + i;
          float x = c[mf][nf][i];
          float part = __shfl_xor(x, 1, 32);
          float f = ropeC[(size_t)row * DH + d];
          float cs = __cosf(f), sn = __sinf(f);
          float val = odd ? (x * cs + part * sn) : (x * cs - part * sn);
          pk[i] = (_Float16)val;
        }
        int bb = r0 >> 10, kvp0 = r0 & (KVL - 1);
        // 8 consecutive kv per lane -> one packed 16B store (free transpose)
        *(v8h*)(Vt + (((size_t)(bb * NHEAD + h) * DH) + d) * KVL + kvp0) = pk;
      }
    }
}

// ---------------------------------------------------------------------------
// 5) Flash attention: one wave per (b,h,16-q tile).  Inverse RoPE fused.
// ---------------------------------------------------------------------------
__global__ __launch_bounds__(128) void attn_kernel(
    const _Float16* __restrict__ Qh, const _Float16* __restrict__ Kh,
    const _Float16* __restrict__ Vt, const float* __restrict__ ropeQ,
    _Float16* __restrict__ AO) {
  __shared__ _Float16 lds[4 * 16 * 40];  // 4 waves x 16x32 P tile, ld=40 (pad)
  int warp = threadIdx.x >> 5, lane = threadIdx.x & 31;
  int wid = blockIdx.x * 4 + warp;
  int bh = wid >> 5;        // (b*8+h)
  int q0 = (wid & 31) * 16; // q tile base within segment
  int b = bh >> 3, h = bh & 7;
  const _Float16* Qbh = Qh + (size_t)bh * QL * DH;
  const _Float16* Kbh = Kh + (size_t)bh * KVL * DH;
  const _Float16* Vbh = Vt + (size_t)bh * DH * KVL;
  _Float16* plds = lds + warp * 640;

  v16h qa0 = load_a_frag(Qbh, DH, q0, 0);
  v16h qa1 = load_a_frag(Qbh, DH, q0, 32);

  v8f o[4];
  const v8f vz = {};
#pragma unroll
  for (int j = 0; j < 4; ++j) o[j] = vz;
  float m[8], l[8];
#pragma unroll
  for (int i = 0; i < 8; ++i) { m[i] = -1e30f; l[i] = 0.f; }

  for (int kv = 0; kv < KVL; kv += 32) {
    v8f s0 = vz, s1 = vz;
    s0 = wmma16(qa0, load_b_frag(Kbh, DH, kv, 0), s0);
    s0 = wmma16(qa1, load_b_frag(Kbh, DH, kv, 32), s0);
    s1 = wmma16(qa0, load_b_frag(Kbh, DH, kv + 16, 0), s1);
    s1 = wmma16(qa1, load_b_frag(Kbh, DH, kv + 16, 32), s1);

    // online softmax per row (rows of a C-frag live within a 16-lane half)
#pragma unroll
    for (int i = 0; i < 8; ++i) {
      float a0 = s0[i] * 0.125f, a1 = s1[i] * 0.125f;
      float rm = fmaxf(a0, a1);
#pragma unroll
      for (int off = 8; off > 0; off >>= 1) rm = fmaxf(rm, __shfl_xor(rm, off, 32));
      float mn = fmaxf(m[i], rm);
      float alpha = __expf(m[i] - mn);
      float p0 = __expf(a0 - mn), p1 = __expf(a1 - mn);
      float rs = p0 + p1;
#pragma unroll
      for (int off = 8; off > 0; off >>= 1) rs += __shfl_xor(rs, off, 32);
      l[i] = l[i] * alpha + rs;
      m[i] = mn;
#pragma unroll
      for (int j = 0; j < 4; ++j) o[j][i] *= alpha;
      int r = ((lane >> 4) << 3) + i;
      plds[r * 40 + (lane & 15)] = (_Float16)p0;       // C-layout -> row-major
      plds[r * 40 + 16 + (lane & 15)] = (_Float16)p1;
    }
    __syncthreads();  // order DS store -> DS load (uniform across all 4 waves)

    // reload P as an A-fragment (transpose via LDS)
    int r = lane & 15, kb = (lane >> 4) << 3;
    v8h lo = *(const v8h*)(plds + r * 40 + kb);
    v8h hi = *(const v8h*)(plds + r * 40 + 16 + kb);
    v16h pa;
#pragma unroll
    for (int t = 0; t < 8; ++t) { pa[t] = lo[t]; pa[t + 8] = hi[t]; }

#pragma unroll
    for (int j = 0; j < 4; ++j)
      o[j] = wmma16(pa, load_b_frag(Vbh, KVL, j * 16, kv), o[j]);
    __syncthreads();
  }

  float inv_l[8];
#pragma unroll
  for (int i = 0; i < 8; ++i) inv_l[i] = 1.0f / l[i];
  int r0 = q0 + ((lane >> 4) << 3);
#pragma unroll
  for (int j = 0; j < 4; ++j) {
    int d = j * 16 + (lane & 15);
    bool odd = d & 1;
#pragma unroll
    for (int i = 0; i < 8; ++i) {
      float x = o[j][i] * inv_l[i];
      float part = __shfl_xor(x, 1, 32);
      int row = r0 + i;
      float f = ropeQ[((size_t)(b * QL + row)) * DH + d];
      float cs = __cosf(f), sn = __sinf(f);
      // inverse rope: even: x*cos + pair*sin ; odd: x*cos - pair*sin
      float val = odd ? (x * cs - part * sn) : (x * cs + part * sn);
      AO[((size_t)(b * QL + row)) * INNERD + h * DH + d] = (_Float16)val;
    }
  }
}

// ---------------------------------------------------------------------------
// 6) Output GEMM + bias -> f32
// ---------------------------------------------------------------------------
__global__ __launch_bounds__(128) void gemm_out_kernel(
    const _Float16* __restrict__ A, const _Float16* __restrict__ Bt,
    const float* __restrict__ bout, float* __restrict__ O) {
  int warp = threadIdx.x >> 5, lane = threadIdx.x & 31;
  int wid = blockIdx.x * 4 + warp;
  int row0 = (wid >> 3) * 64, col0 = (wid & 7) * 64;
  v8f c[4][4];
  const v8f vz = {};
#pragma unroll
  for (int i = 0; i < 4; ++i)
#pragma unroll
    for (int j = 0; j < 4; ++j) c[i][j] = vz;
  gemm_core(A, Bt, row0, col0, INNERD, c);

#pragma unroll
  for (int mf = 0; mf < 4; ++mf)
#pragma unroll
    for (int nf = 0; nf < 4; ++nf) {
      int n = col0 + nf * 16 + (lane & 15);
      float bo = bout[n];
      int r0 = row0 + mf * 16 + ((lane >> 4) << 3);
#pragma unroll
      for (int i = 0; i < 8; ++i)
        O[(size_t)(r0 + i) * DIMM + n] = c[mf][nf][i] + bo;
    }
}

// ---------------------------------------------------------------------------
extern "C" void kernel_launch(void* const* d_in, const int* in_sizes, int n_in,
                              void* d_out, int out_size, void* d_ws, size_t ws_size,
                              hipStream_t stream) {
  const float* x_query   = (const float*)d_in[0];
  const float* x_context = (const float*)d_in[1];
  const float* ropeQ     = (const float*)d_in[2];
  const float* ropeC     = (const float*)d_in[3];
  const float* ln_q_g    = (const float*)d_in[4];
  const float* ln_q_b    = (const float*)d_in[5];
  const float* ln_c_g    = (const float*)d_in[6];
  const float* ln_c_b    = (const float*)d_in[7];
  const float* Wq        = (const float*)d_in[8];
  const float* Wkv       = (const float*)d_in[9];
  const float* Wout      = (const float*)d_in[10];
  const float* bout      = (const float*)d_in[11];
  float* out = (float*)d_out;

  char* ws = (char*)d_ws;
  _Float16* xqn   = (_Float16*)ws; ws += (size_t)NQ_TOK * DIMM * 2;
  _Float16* xcn   = (_Float16*)ws; ws += (size_t)NKV_TOK * DIMM * 2;
  _Float16* WqT   = (_Float16*)ws; ws += (size_t)INNERD * DIMM * 2;
  _Float16* WkvT  = (_Float16*)ws; ws += (size_t)(2 * INNERD) * DIMM * 2;
  _Float16* WoutT = (_Float16*)ws; ws += (size_t)DIMM * INNERD * 2;
  _Float16* Qh    = (_Float16*)ws; ws += (size_t)NQ_TOK * DH * NHEAD * 2;
  _Float16* Kh    = (_Float16*)ws; ws += (size_t)NKV_TOK * DH * NHEAD * 2;
  _Float16* Vt    = (_Float16*)ws; ws += (size_t)NKV_TOK * DH * NHEAD * 2;
  _Float16* AO    = (_Float16*)ws; ws += (size_t)NQ_TOK * INNERD * 2;

  ln_f16_kernel<<<NQ_TOK / 8, 256, 0, stream>>>(x_query, ln_q_g, ln_q_b, xqn, NQ_TOK);
  ln_f16_kernel<<<NKV_TOK / 8, 256, 0, stream>>>(x_context, ln_c_g, ln_c_b, xcn, NKV_TOK);

  wt_f16_kernel<<<(INNERD * DIMM) / 256, 256, 0, stream>>>(Wq, WqT, INNERD, DIMM);
  wt_f16_kernel<<<(2 * INNERD * DIMM) / 256, 256, 0, stream>>>(Wkv, WkvT, 2 * INNERD, DIMM);
  wt_f16_kernel<<<(DIMM * INNERD) / 256, 256, 0, stream>>>(Wout, WoutT, DIMM, INNERD);

  // (8192/64)*(512/64)=1024 waves, 4 per block
  gemm_q_kernel<<<256, 128, 0, stream>>>(xqn, WqT, ropeQ, Qh);
  // (16384/64)*(1024/64)=4096 waves
  gemm_kv_kernel<<<1024, 128, 0, stream>>>(xcn, WkvT, ropeC, Kh, Vt);
  // 16*8*(512/16)=4096 waves
  attn_kernel<<<1024, 128, 0, stream>>>(Qh, Kh, Vt, ropeQ, AO);
  // 1024 waves
  gemm_out_kernel<<<256, 128, 0, stream>>>(AO, WoutT, bout, out);
}
